// RGCNEncoder_52407190946326
// MI455X (gfx1250) — matrix operand order repaired and checked by reading, not compiled, not run on hardware
//
#include <hip/hip_runtime.h>
#include <hip/hip_bf16.h>

#define NN     50000
#define HID    128
#define NREL   9
#define NLAY   3
#define NEDGE  600000
#define MTILES (NN / 16)        // 3125, exact

typedef __attribute__((ext_vector_type(16))) __bf16 v16bf;
typedef __attribute__((ext_vector_type(8)))  __bf16 v8bf;
typedef __attribute__((ext_vector_type(8)))  float  v8f;

// ---------------- utility kernels ----------------

__global__ void zero_f32(float* __restrict__ p, int n) {
    for (int i = blockIdx.x * blockDim.x + threadIdx.x; i < n; i += gridDim.x * blockDim.x)
        p[i] = 0.0f;
}

__global__ void relu_f32(float* __restrict__ p, int n) {
    for (int i = blockIdx.x * blockDim.x + threadIdx.x; i < n; i += gridDim.x * blockDim.x) {
        float v = p[i];
        p[i] = v > 0.0f ? v : 0.0f;
    }
}

__global__ void cvt_f32_bf16(const float* __restrict__ in, __bf16* __restrict__ out, int n) {
    for (int i = blockIdx.x * blockDim.x + threadIdx.x; i < n; i += gridDim.x * blockDim.x)
        out[i] = (__bf16)in[i];
}

// ---------------- WMMA GEMM: X_r = Hb (N x 128, bf16) @ Wr (128 x 128, bf16) ----------------
// block = 256 threads = 8 waves. blockIdx.y = 16-wide feature tile (0..7),
// blockIdx.x*8 + wave = 16-row M tile. One v_wmma_f32_16x16x32_bf16 tile per wave,
// K-loop of 4 over HID=128. B panel (128 x 16) staged in LDS, padded to avoid
// bank conflicts (stride 136 halves = 68 dwords -> lane n hits bank 4n).

__global__ void __launch_bounds__(256)
rgcn_gemm_bf16(const __bf16* __restrict__ Hb,
               const __bf16* __restrict__ Wr,   // [128][128] this relation's weight
               float* __restrict__ Xr) {
    __shared__ __align__(16) __bf16 Wl[16 * 136];

    const int tid   = threadIdx.x;
    const int wave  = tid >> 5;
    const int lane  = tid & 31;
    const int ntile = blockIdx.y;                 // feature tile 0..7
    const int mtile = blockIdx.x * 8 + wave;      // 16-row tile

    // Cooperative load of W columns [ntile*16 .. ntile*16+15], all K, transposed:
    // Wl[c*136 + k] = Wr[k*HID + ntile*16 + c]
    for (int i = tid; i < 16 * HID; i += 256) {
        const int c = i >> 7;       // 0..15
        const int k = i & 127;      // 0..127
        Wl[c * 136 + k] = Wr[k * HID + ntile * 16 + c];
    }
    __syncthreads();

    if (mtile >= MTILES) return;    // wave-uniform -> EXEC stays all-ones for WMMA

    const int mrow  = mtile * 16 + (lane & 15);   // A row for this lane
    const int nkoff = lane >> 4;                  // 0 or 1 (half-wave)
    const int ncol  = lane & 15;                  // B/D column within tile

    v8f acc = {};
#pragma unroll
    for (int ks = 0; ks < 4; ++ks) {
        // A fragment: elements 0..7 = K0..K0+7, elements 8..15 = K0+16..K0+23,
        // K0 = ks*32 + nkoff*8  (per CDNA5 16-bit A layout)
        const int k0a = ks * 32 + nkoff * 8;
        const v8bf* ap = (const v8bf*)(Hb + (size_t)mrow * HID + k0a);
        const v8bf alo = ap[0];
        const v8bf ahi = ap[2];     // +16 halves
        v16bf a;
#pragma unroll
        for (int i = 0; i < 8; ++i) { a[i] = alo[i]; a[8 + i] = ahi[i]; }

        // B fragment: lane holds column ncol, elements 0..7 = K0..K0+7, 8..15 = K0+16..K0+23, K0 = ks*32 + nkoff*8
        const int kb = ks * 32 + nkoff * 8;
        const v8bf* bp = (const v8bf*)(&Wl[ncol * 136 + kb]);
        const v8bf blo = bp[0];
        const v8bf bhi = bp[2];     // +16 halves
        v16bf b;
#pragma unroll
        for (int i = 0; i < 8; ++i) { b[i] = blo[i]; b[8 + i] = bhi[i]; }

        acc = __builtin_amdgcn_wmma_f32_16x16x32_bf16(false, a, false, b,
                                                      (short)0, acc, false, false);
    }

    // D layout: VGPR j holds M = mtile*16 + nkoff*8 + j, column = ntile*16 + ncol
    const int ng    = ntile * 16 + ncol;
    const int mbase = mtile * 16 + nkoff * 8;
#pragma unroll
    for (int j = 0; j < 8; ++j)
        Xr[(size_t)(mbase + j) * HID + ng] = acc[j];
}

// ---------------- edge gather + bias + scatter-add (one relation) ----------------
// 32 threads (1 wave) per edge; each thread owns a float4 chunk of the 128-wide row.
// X_r is L2-resident (25.6 MB << 192 MB), so the random gather hits L2.

__global__ void __launch_bounds__(256)
rgcn_edge_scatter(const int* __restrict__ ei,    // [2][NEDGE]: row 0 = dest, row 1 = src
                  const int* __restrict__ et,    // [NEDGE]
                  const float* __restrict__ Xr,  // [NN][128]
                  const float* __restrict__ bias,// [128] for (layer, rel)
                  float* __restrict__ acc,       // [NN][128]
                  int rel) {
    const int gid = blockIdx.x * blockDim.x + threadIdx.x;
    const int e = gid >> 5;
    if (e >= NEDGE) return;
    if (et[e] != rel) return;

    const int lane4 = (gid & 31) * 4;
    const int dest = ei[e];
    const int src  = ei[NEDGE + e];

    const float4 x  = *(const float4*)(Xr + (size_t)src * HID + lane4);
    const float4 bb = *(const float4*)(bias + lane4);
    float* p = acc + (size_t)dest * HID + lane4;
    atomicAdd(p + 0, x.x + bb.x);
    atomicAdd(p + 1, x.y + bb.y);
    atomicAdd(p + 2, x.z + bb.z);
    atomicAdd(p + 3, x.w + bb.w);
}

// ---------------- launch ----------------

extern "C" void kernel_launch(void* const* d_in, const int* in_sizes, int n_in,
                              void* d_out, int out_size, void* d_ws, size_t ws_size,
                              hipStream_t stream) {
    (void)in_sizes; (void)n_in; (void)out_size; (void)ws_size;

    const int*   ei  = (const int*)d_in[0];    // [2, NEDGE]
    const int*   et  = (const int*)d_in[1];    // [NEDGE]
    const float* emb = (const float*)d_in[2];  // [NN, HID]
    const float* wts = (const float*)d_in[3];  // [NLAY, NREL, HID, HID]
    const float* bs  = (const float*)d_in[4];  // [NLAY, NREL, HID]
    float*       out = (float*)d_out;          // [NN, HID]

    // workspace partition (all chunks 256B-multiples)
    char* ws = (char*)d_ws;
    __bf16* Hb  = (__bf16*)ws;                 ws += (size_t)NN * HID * 2;        // 12.8 MB
    __bf16* Wb  = (__bf16*)ws;                 ws += (size_t)NREL * HID * HID * 2;// 0.29 MB
    float*  Xr  = (float*)ws;                  ws += (size_t)NN * HID * 4;        // 25.6 MB
    float*  buf0 = (float*)ws;                 ws += (size_t)NN * HID * 4;        // 25.6 MB
    float*  buf1 = (float*)ws;                                                   // 25.6 MB

    const int nfeat = NN * HID;                    // 6.4M
    const dim3 gGemm((MTILES + 7) / 8, 8, 1);      // 391 x 8 blocks, 256 thr
    const int  eBlocks = (NEDGE * 32 + 255) / 256; // 75000 blocks

    const float* cur = emb;
    for (int l = 0; l < NLAY; ++l) {
        float* nxt = (l == 0) ? buf0 : (l == 1) ? buf1 : out;

        cvt_f32_bf16<<<4096, 256, 0, stream>>>(cur, Hb, nfeat);
        cvt_f32_bf16<<<1024, 256, 0, stream>>>(wts + (size_t)l * NREL * HID * HID,
                                               Wb, NREL * HID * HID);
        zero_f32<<<4096, 256, 0, stream>>>(nxt, nfeat);

        for (int r = 0; r < NREL; ++r) {
            rgcn_gemm_bf16<<<gGemm, 256, 0, stream>>>(Hb, Wb + (size_t)r * HID * HID, Xr);
            rgcn_edge_scatter<<<eBlocks, 256, 0, stream>>>(
                ei, et, Xr, bs + (size_t)(l * NREL + r) * HID, nxt, r);
        }

        if (l + 1 < NLAY)
            relu_f32<<<4096, 256, 0, stream>>>(nxt, nfeat);
        cur = nxt;
    }
}
